// BSCLossSingleLabel_2697239462642
// MI455X (gfx1250) — matrix operand order repaired and checked by reading, not compiled, not run on hardware
//
#include <hip/hip_runtime.h>
#include <hip/hip_bf16.h>

#define BATCH   8192
#define HIDDEN  256
#define NUMC    32
#define INV_T   10.0f          // 1 / TEMPERATURE

typedef __attribute__((ext_vector_type(16))) _Float16 v16h;
typedef __attribute__((ext_vector_type(8)))  _Float16 v8h;
typedef __attribute__((ext_vector_type(8)))  float    v8f;

// Workspace layout (floats):
//  wsf[0] = loss sum,  wsf[1] = valid count
//  ((int*)(wsf+2))[0..31] = class counts
//  fnorm (f16, BATCH*HIDDEN) at byte offset 256
static __device__ __forceinline__ _Float16* ws_fnorm(void* ws) {
    return (_Float16*)((char*)ws + 256);
}

// ---------------------------------------------------------------------------
// Kernel 0: zero the small accumulators
// ---------------------------------------------------------------------------
__global__ void k_init(float* wsf) {
    int t = threadIdx.x;
    if (t < 2) wsf[t] = 0.0f;
    int* cc = (int*)(wsf + 2);
    if (t < NUMC) cc[t] = 0;
}

// ---------------------------------------------------------------------------
// Kernel 1: L2-normalize rows (f32 -> f16) + label histogram
// One wave (32 lanes) per row; 8 rows per 256-thread block.
// ---------------------------------------------------------------------------
__global__ void k_normalize(const float* __restrict__ feat,
                            const int*   __restrict__ labels,
                            float* __restrict__ wsf) {
    const int lane = threadIdx.x & 31;
    const int wave = threadIdx.x >> 5;
    const int row  = blockIdx.x * 8 + wave;
    if (row >= BATCH) return;

    const float* rp = feat + (size_t)row * HIDDEN;
    float v[8];
    float ss = 0.0f;
#pragma unroll
    for (int j = 0; j < 8; ++j) {
        v[j] = rp[lane + j * 32];
        ss += v[j] * v[j];
    }
#pragma unroll
    for (int off = 16; off > 0; off >>= 1)
        ss += __shfl_xor(ss, off, 32);

    const float inv = rsqrtf(ss);
    _Float16* op = ws_fnorm((void*)wsf) + (size_t)row * HIDDEN;
#pragma unroll
    for (int j = 0; j < 8; ++j)
        op[lane + j * 32] = (_Float16)(v[j] * inv);

    if (lane == 0) {
        int* cc = (int*)(wsf + 2);
        atomicAdd(&cc[labels[row]], 1);
    }
}

// ---------------------------------------------------------------------------
// Kernel 2: fused sim-GEMM (WMMA f16->f32) + exp/class-sum/pos-sum epilogue
// + per-anchor loss. Grid = BATCH/16 workgroups of 256 threads (8 waves).
// Each workgroup owns 16 anchor rows; each wave processes TWO 16-column
// tiles per iteration (dual independent WMMA accumulator chains for XDL
// pipelining and A-fragment reuse).
// ---------------------------------------------------------------------------
__global__ void k_scl_fused(const int* __restrict__ labels,
                            float* __restrict__ wsf) {
    __shared__ float sClassSum[16][NUMC];
    __shared__ float sPosSum[16];

    const int tid     = threadIdx.x;
    const int lane    = tid & 31;
    const int waveId  = tid >> 5;
    const int laneHi  = lane >> 4;       // 0/1 half-wave
    const int nCol    = lane & 15;       // output column within tile
    const int rowBase = blockIdx.x * 16;

    // zero LDS accumulators
    for (int i = tid; i < 16 * NUMC; i += 256) ((float*)sClassSum)[i] = 0.0f;
    if (tid < 16) sPosSum[tid] = 0.0f;
    __syncthreads();

    const _Float16* fnorm = ws_fnorm((void*)wsf);
    const int* cc = (const int*)(wsf + 2);

    // Preload A fragments: A row = rowBase + nCol (lanes 0-15 mirror 16-31).
    // ISA 16-bit A 16x32 layout: lane holds K = {0..7}+8*hi and {16..23}+8*hi.
    const _Float16* arow = fnorm + (size_t)(rowBase + nCol) * HIDDEN;
    v16h aFrag[8];
#pragma unroll
    for (int kt = 0; kt < 8; ++kt) {
        v8h lo = *(const v8h*)(arow + kt * 32 + laneHi * 8);
        v8h hi = *(const v8h*)(arow + kt * 32 + 16 + laneHi * 8);
        v16h a;
#pragma unroll
        for (int i = 0; i < 8; ++i) { a[i] = lo[i]; a[i + 8] = hi[i]; }
        aFrag[kt] = a;
    }

    // Anchor labels for this lane's half (C rows m = r + 8*laneHi)
    int aLab[8];
#pragma unroll
    for (int r = 0; r < 8; ++r)
        aLab[r] = labels[rowBase + r + laneHi * 8];

    const int ITERS = BATCH / 256;       // 8 waves * 32 cols = 256 cols/iter
    for (int it = 0; it < ITERS; ++it) {
        const int nGlob0 = it * 256 + waveId * 32 + nCol;
        const int nGlob1 = nGlob0 + 16;
        const _Float16* brow0 = fnorm + (size_t)nGlob0 * HIDDEN;
        const _Float16* brow1 = fnorm + (size_t)nGlob1 * HIDDEN;

        // issue label loads early; latency hidden under the WMMA chain
        const int cCol0 = labels[nGlob0];
        const int cCol1 = labels[nGlob1];

        v8f acc0 = {};
        v8f acc1 = {};
#pragma unroll
        for (int kt = 0; kt < 8; ++kt) {
            // ISA 16-bit B 32x16 layout: lane = column, K = {0..7}+8*hi and {16..23}+8*hi
            v8h b0lo = *(const v8h*)(brow0 + kt * 32 + laneHi * 8);
            v8h b0hi = *(const v8h*)(brow0 + kt * 32 + 16 + laneHi * 8);
            v8h b1lo = *(const v8h*)(brow1 + kt * 32 + laneHi * 8);
            v8h b1hi = *(const v8h*)(brow1 + kt * 32 + 16 + laneHi * 8);
            v16h b0, b1;
#pragma unroll
            for (int i = 0; i < 8; ++i) { b0[i] = b0lo[i]; b0[i + 8] = b0hi[i]; b1[i] = b1lo[i]; b1[i + 8] = b1hi[i]; }
            acc0 = __builtin_amdgcn_wmma_f32_16x16x32_f16(
                false, aFrag[kt], false, b0, (short)0, acc0, false, false);
            acc1 = __builtin_amdgcn_wmma_f32_16x16x32_f16(
                false, aFrag[kt], false, b1, (short)0, acc1, false, false);
        }

#pragma unroll
        for (int r = 0; r < 8; ++r) {
            const int mLoc  = r + laneHi * 8;
            const int mGlob = rowBase + mLoc;
            const float sim0 = acc0[r] * INV_T;
            const float sim1 = acc1[r] * INV_T;
            // diagonal: contribute exactly 0 to class sums (branchless),
            // keep exec-masking only for the rare positive-pair adds
            const float e0 = (mGlob != nGlob0) ? __expf(sim0) : 0.0f;
            const float e1 = (mGlob != nGlob1) ? __expf(sim1) : 0.0f;
            atomicAdd(&sClassSum[mLoc][cCol0], e0);
            atomicAdd(&sClassSum[mLoc][cCol1], e1);
            if (cCol0 == aLab[r] && mGlob != nGlob0)
                atomicAdd(&sPosSum[mLoc], sim0);
            if (cCol1 == aLab[r] && mGlob != nGlob1)
                atomicAdd(&sPosSum[mLoc], sim1);
        }
    }
    __syncthreads();

    // Per-anchor loss, reduced to global accumulators
    if (tid < 16) {
        const int m = tid;
        const int lab = labels[rowBase + m];
        float denom = 0.0f;
#pragma unroll
        for (int c = 0; c < NUMC; ++c) {
            int cnt = cc[c] - (c == lab ? 1 : 0);
            if (cnt > 0) denom += sClassSum[m][c] / (float)cnt;
        }
        const int P = cc[lab] - 1;
        if (P > 0) {
            float li = __logf(fmaxf(denom, 1e-30f)) - sPosSum[m] / (float)P;
            atomicAdd(&wsf[0], li);
            atomicAdd(&wsf[1], 1.0f);
        }
    }
}

// ---------------------------------------------------------------------------
// Kernel 3: finalize
// ---------------------------------------------------------------------------
__global__ void k_finalize(const float* __restrict__ wsf,
                           float* __restrict__ out) {
    const float s = wsf[0];
    const float v = wsf[1];
    out[0] = (v > 0.0f) ? (s / v) : 0.0f;
}

// ---------------------------------------------------------------------------
extern "C" void kernel_launch(void* const* d_in, const int* in_sizes, int n_in,
                              void* d_out, int out_size, void* d_ws, size_t ws_size,
                              hipStream_t stream) {
    const float* feat   = (const float*)d_in[0];
    const int*   labels = (const int*)d_in[1];
    float*       wsf    = (float*)d_ws;
    float*       out    = (float*)d_out;

    k_init<<<1, 64, 0, stream>>>(wsf);
    k_normalize<<<BATCH / 8, 256, 0, stream>>>(feat, labels, wsf);
    k_scl_fused<<<BATCH / 16, 256, 0, stream>>>(labels, wsf);
    k_finalize<<<1, 1, 0, stream>>>(wsf, out);
}
